// GraphAttentionLayer_rel_36825049595917
// MI455X (gfx1250) — matrix operand order repaired and checked by reading, not compiled, not run on hardware
//
#include <hip/hip_runtime.h>
#include <hip/hip_bf16.h>
#include <cmath>

typedef __attribute__((ext_vector_type(16))) _Float16 v16h;
typedef __attribute__((ext_vector_type(8)))  float    v8f;

static constexpr int    NN    = 10000;
static constexpr int    EE    = 200000;
static constexpr int    RR    = 400000;
static constexpr int    FIN   = 128;
static constexpr int    FOUT  = 64;
static constexpr int    KPAD  = 10016;          // 313 * 32
static constexpr int    KB    = KPAD / 32;      // 313 K-blocks
static constexpr int    MT    = NN / 16;        // 625 row tiles

// P stored fragment-major: [tile][kb][lane(32)][16 halfs] -> v16h index (it*KB+kb)*32+lane
static constexpr size_t PFRAG_BYTES = (size_t)MT * KB * 32 * 16 * 2;   // 200,320,000
static constexpr size_t SUMS_OFF    = PFRAG_BYTES;
static constexpr size_t CNTS_OFF    = SUMS_OFF + (size_t)EE * 4;
static constexpr size_t SEQT_OFF    = CNTS_OFF + (size_t)EE * 4;       // 32B-aligned
static constexpr size_t Z_OFF       = SEQT_OFF + (size_t)FOUT * KPAD * 2;

__device__ __forceinline__ v8f wmma16x16x32(v16h a, v16h b, v8f c) {
  return __builtin_amdgcn_wmma_f32_16x16x32_f16(false, a, false, b, (short)0, c,
                                                false, false);
}

// ---------------- K0: zero segment sums + counts (contiguous) ----------------
__global__ void k_zero(float* p, long long n) {
  long long i = (long long)blockIdx.x * blockDim.x + threadIdx.x;
  if (i < n) p[i] = 0.0f;
}

// ---------------- K1: fts_rel = rel . w_rel, segment-sum via atomics --------
__global__ __launch_bounds__(256) void k_relmv(const float* __restrict__ rel,
                                               const float* __restrict__ wrel,
                                               const int* __restrict__ seg,
                                               float* sums, float* cnts) {
  int wid  = blockIdx.x * 8 + (threadIdx.x >> 5);   // one wave32 per rel row
  int lane = threadIdx.x & 31;
  if (wid >= RR) return;
  float4 r4 = ((const float4*)(rel + (size_t)wid * FIN))[lane];
  float4 w4 = ((const float4*)wrel)[lane];
  float p = r4.x * w4.x + r4.y * w4.y + r4.z * w4.z + r4.w * w4.w;
  #pragma unroll
  for (int off = 16; off; off >>= 1) p += __shfl_down(p, off, 32);
  if (lane == 0) {
    int s = seg[wid];
    atomicAdd(&sums[s], p);
    atomicAdd(&cnts[s], 1.0f);
  }
}

// ---------------- K2: edge_means = sums / max(cnts,1) (in place) ------------
__global__ void k_means(float* sums, const float* cnts) {
  int e = blockIdx.x * blockDim.x + threadIdx.x;
  if (e < EE) sums[e] = sums[e] / fmaxf(cnts[e], 1.0f);
}

// ---------------- K3: P = exp(adj + 0.5) as f16 A-fragments, Z = row sums ---
// One block (8 waves) per 16-row tile.  Lane l owns row (l&15); its v16h
// fragment covers k = k0 + {khalf*8..+7, 16+khalf*8..+7}  (ISA 7.12.2 layout).
__global__ __launch_bounds__(256) void k_pbuild(const float* __restrict__ adj,
                                                _Float16* __restrict__ Pfrag,
                                                float* __restrict__ Z) {
  __shared__ float zred[8][16];
  const int tid = threadIdx.x, w = tid >> 5, lane = tid & 31;
  const int it = blockIdx.x, i0 = it * 16;
  const int r = lane & 15, khalf = lane >> 4;
  const float* arow = adj + (size_t)(i0 + r) * NN;
  v16h* pfr = (v16h*)Pfrag + (size_t)it * KB * 32 + lane;
  float zacc = 0.0f;

  for (int kb = w; kb < KB; kb += 8) {
    int k0 = kb * 32;
    int k1 = k0 + khalf * 8;        // group 1: always < NN (max 9992)
    int k2 = k0 + 16 + khalf * 8;   // group 2: invalid only in last kb
    v16h a;
    {
      float4 u = *(const float4*)(arow + k1);
      float4 v = *(const float4*)(arow + k1 + 4);
      float e0 = __expf(u.x + 0.5f), e1 = __expf(u.y + 0.5f);
      float e2 = __expf(u.z + 0.5f), e3 = __expf(u.w + 0.5f);
      float e4 = __expf(v.x + 0.5f), e5 = __expf(v.y + 0.5f);
      float e6 = __expf(v.z + 0.5f), e7 = __expf(v.w + 0.5f);
      zacc += (e0 + e1 + e2 + e3) + (e4 + e5 + e6 + e7);
      a[0] = (_Float16)e0; a[1] = (_Float16)e1; a[2] = (_Float16)e2;
      a[3] = (_Float16)e3; a[4] = (_Float16)e4; a[5] = (_Float16)e5;
      a[6] = (_Float16)e6; a[7] = (_Float16)e7;
    }
    if (k2 < NN) {
      float4 u = *(const float4*)(arow + k2);
      float4 v = *(const float4*)(arow + k2 + 4);
      float e0 = __expf(u.x + 0.5f), e1 = __expf(u.y + 0.5f);
      float e2 = __expf(u.z + 0.5f), e3 = __expf(u.w + 0.5f);
      float e4 = __expf(v.x + 0.5f), e5 = __expf(v.y + 0.5f);
      float e6 = __expf(v.z + 0.5f), e7 = __expf(v.w + 0.5f);
      zacc += (e0 + e1 + e2 + e3) + (e4 + e5 + e6 + e7);
      a[8]  = (_Float16)e0; a[9]  = (_Float16)e1; a[10] = (_Float16)e2;
      a[11] = (_Float16)e3; a[12] = (_Float16)e4; a[13] = (_Float16)e5;
      a[14] = (_Float16)e6; a[15] = (_Float16)e7;
    } else {
      #pragma unroll
      for (int i = 8; i < 16; ++i) a[i] = (_Float16)0.0f;
    }
    pfr[(size_t)kb * 32] = a;       // lane-major: 1KB contiguous per wave
  }
  // lanes l and l+16 hold the same row -> fold, then reduce across 8 waves
  zacc += __shfl_down(zacc, 16, 32);
  if (lane < 16) zred[w][lane] = zacc;
  __syncthreads();
  if (tid < 16) {
    float z = 0.0f;
    #pragma unroll
    for (int w2 = 0; w2 < 8; ++w2) z += zred[w2][tid];
    Z[i0 + tid] = z;
  }
}

// ---------------- K4: scatter edge corrections into P fragments + fix Z -----
__device__ __forceinline__ void p_correct(int i, int j, float em_sig,
                                          const float* adj, _Float16* Pfrag,
                                          float* Z) {
  float base = adj[(size_t)i * NN + j];
  float vnew = __expf(base + em_sig);
  float vold = __expf(base + 0.5f);
  int it = i >> 4, r = i & 15, kb = j >> 5, kk = j & 31;
  int seg = kk >> 3, sub = kk & 7;
  int lane = r + ((seg & 1) << 4);
  int idx  = ((seg >> 1) << 3) + sub;
  size_t addr = (((size_t)it * KB + kb) * 32 + lane) * 16 + idx;
  Pfrag[addr] = (_Float16)vnew;
  atomicAdd(&Z[i], vnew - vold);
}

__global__ void k_pscatter(const int* __restrict__ ep, const float* __restrict__ em,
                           const float* __restrict__ adj, _Float16* Pfrag,
                           float* Z) {
  int e = blockIdx.x * blockDim.x + threadIdx.x;
  if (e >= EE) return;
  int e1 = ep[2 * e], e2 = ep[2 * e + 1];
  float sg = 1.0f / (1.0f + __expf(-em[e]));
  p_correct(e1, e2, sg, adj, Pfrag, Z);
  p_correct(e2, e1, sg, adj, Pfrag, Z);
}

// ---------------- K5: seq_fts = x @ W^T via WMMA, stored transposed f16 -----
__global__ __launch_bounds__(32) void k_seqfts(const float* __restrict__ x,
                                               const float* __restrict__ W,
                                               _Float16* __restrict__ seqT) {
  __shared__ _Float16 xt[16 * 32];
  const int lane = threadIdx.x;
  if (blockIdx.x == MT) {                 // zero the K-padding columns
    for (int t = lane; t < FOUT * 16; t += 32) {
      int n = t >> 4, k = NN + (t & 15);
      seqT[(size_t)n * KPAD + k] = (_Float16)0.0f;
    }
    return;
  }
  const int i0   = blockIdx.x * 16;
  const int prow = lane >> 1, pcol0 = (lane & 1) * 16;
  const float* xrow = x + (size_t)(i0 + prow) * FIN + pcol0;
  const int rowa = lane & 15, khalf = lane >> 4;
  v8f c[4];
  #pragma unroll
  for (int t = 0; t < 4; ++t)
    #pragma unroll
    for (int u = 0; u < 8; ++u) c[t][u] = 0.0f;

  for (int kb = 0; kb < FIN / 32; ++kb) {
    int k0 = kb * 32;
    #pragma unroll
    for (int q = 0; q < 16; ++q)
      xt[prow * 32 + pcol0 + q] = (_Float16)xrow[k0 + q];
    asm volatile("s_wait_dscnt 0" ::: "memory");
    v16h a;
    #pragma unroll
    for (int i = 0; i < 8; ++i) {
      a[i]     = xt[rowa * 32 + khalf * 8 + i];
      a[i + 8] = xt[rowa * 32 + 16 + khalf * 8 + i];
    }
    #pragma unroll
    for (int nt = 0; nt < 4; ++nt) {                 // B[k][n] = W[n][k]
      int n = nt * 16 + rowa;
      const float* wp = W + (size_t)n * FIN + k0 + khalf * 8;
      v16h b;
      #pragma unroll
      for (int i = 0; i < 16; ++i) b[i] = (_Float16)wp[(i & 7) + ((i >> 3) << 4)];
      c[nt] = wmma16x16x32(a, b, c[nt]);
    }
  }
  #pragma unroll
  for (int nt = 0; nt < 4; ++nt)
    #pragma unroll
    for (int vi = 0; vi < 8; ++vi) {
      int m = vi + khalf * 8, n = nt * 16 + rowa;
      seqT[(size_t)n * KPAD + i0 + m] = (_Float16)c[nt][vi];
    }
}

// ---------------- K6: GEMM  out = (P @ seq_fts)/Z + bias, elu ---------------
// 8 waves split K; A fragments load as one coalesced v16h per lane.
__global__ __launch_bounds__(256) void k_gemm(const _Float16* __restrict__ Pfrag,
                                              const _Float16* __restrict__ seqT,
                                              const float* __restrict__ Z,
                                              const float* __restrict__ bias,
                                              float* __restrict__ out) {
  __shared__ float red[8][16][64];
  const int tid = threadIdx.x, w = tid >> 5, lane = tid & 31;
  const int i0 = blockIdx.x * 16;
  const int rowa = lane & 15, khalf = lane >> 4;
  const v16h* aptr = (const v16h*)Pfrag + (size_t)blockIdx.x * KB * 32 + lane;
  v8f c[4];
  #pragma unroll
  for (int t = 0; t < 4; ++t)
    #pragma unroll
    for (int u = 0; u < 8; ++u) c[t][u] = 0.0f;

  for (int kb = w; kb < KB; kb += 8) {
    int k0 = kb * 32;
    v16h a = aptr[(size_t)kb * 32];
    if (kb + 8 < KB) __builtin_prefetch(aptr + (size_t)(kb + 8) * 32, 0, 0);
    #pragma unroll
    for (int nt = 0; nt < 4; ++nt) {      // B contiguous from transposed seqT
      const _Float16* bp = seqT + (size_t)(nt * 16 + rowa) * KPAD +
                                     k0 + khalf * 8;
      v16h b;
      #pragma unroll
      for (int i = 0; i < 8; ++i) { b[i] = bp[i]; b[i + 8] = bp[16 + i]; }
      c[nt] = wmma16x16x32(a, b, c[nt]);
    }
  }

  #pragma unroll
  for (int nt = 0; nt < 4; ++nt)
    #pragma unroll
    for (int vi = 0; vi < 8; ++vi)
      red[w][vi + khalf * 8][nt * 16 + rowa] = c[nt][vi];
  __syncthreads();

  for (int o = tid; o < 16 * 64; o += 256) {
    int m = o >> 6, n = o & 63;
    float acc = 0.0f;
    #pragma unroll
    for (int w2 = 0; w2 < 8; ++w2) acc += red[w2][m][n];
    int row = i0 + m;
    float val = acc / Z[row] + bias[n];
    out[(size_t)row * FOUT + n] = (val > 0.0f) ? val : expm1f(val);  // elu
  }
}

extern "C" void kernel_launch(void* const* d_in, const int* in_sizes, int n_in,
                              void* d_out, int out_size, void* d_ws, size_t ws_size,
                              hipStream_t stream) {
  const float* x       = (const float*)d_in[0];
  const float* rel     = (const float*)d_in[1];
  const float* adj     = (const float*)d_in[2];
  const int*   edges   = (const int*)d_in[3];
  const int*   rel_seg = (const int*)d_in[4];
  const float* W       = (const float*)d_in[5];
  const float* w_rel   = (const float*)d_in[6];
  const float* bias    = (const float*)d_in[7];
  float*       out     = (float*)d_out;

  char* ws = (char*)d_ws;
  _Float16* Pfrag = (_Float16*)(ws + 0);
  float*    sums  = (float*)(ws + SUMS_OFF);
  float*    cnts  = (float*)(ws + CNTS_OFF);
  _Float16* seqT  = (_Float16*)(ws + SEQT_OFF);
  float*    Z     = (float*)(ws + Z_OFF);

  k_zero<<<(2 * EE + 255) / 256, 256, 0, stream>>>(sums, 2LL * EE);
  k_relmv<<<RR / 8, 256, 0, stream>>>(rel, w_rel, rel_seg, sums, cnts);
  k_means<<<(EE + 255) / 256, 256, 0, stream>>>(sums, cnts);

  k_pbuild<<<MT, 256, 0, stream>>>(adj, Pfrag, Z);
  k_pscatter<<<(EE + 255) / 256, 256, 0, stream>>>(edges, sums, adj, Pfrag, Z);

  k_seqfts<<<MT + 1, 32, 0, stream>>>(x, W, seqT);
  k_gemm<<<MT, 256, 0, stream>>>(Pfrag, seqT, Z, bias, out);
}
